// ZipfWeightLearner_90632399880321
// MI455X (gfx1250) — compile-verified, matrix operand order checked by reading, not run on hardware
//
#include <hip/hip_runtime.h>
#include <hip/hip_bf16.h>
#include <math.h>

// Problem sizes (fixed by reference)
#define B_ 128
#define V_ 1024
#define D_ 512

typedef __attribute__((ext_vector_type(16))) _Float16 v16h;
typedef __attribute__((ext_vector_type(8)))  float    v8f;
typedef __attribute__((ext_vector_type(4)))  int      v4i;

// ---------------------------------------------------------------------------
// Async global -> LDS copy (CDNA5 GLOBAL_LOAD_ASYNC_TO_LDS_B128, ASYNCcnt).
// Builtin signature (from hipcc diagnostic): arg0 = int4 AS1* (global src),
// arg1 = LDS dst, arg2 = imm offset, arg3 = imm cpol.
// Guarded: falls back to load+ds_store if the builtin is absent.
// ---------------------------------------------------------------------------
#if __has_builtin(__builtin_amdgcn_global_load_async_to_lds_b128) && \
    __has_builtin(__builtin_amdgcn_s_wait_asynccnt)
#define HAVE_ASYNC_LDS 1
#else
#define HAVE_ASYNC_LDS 0
#endif

__device__ __forceinline__ void lds_copy16(float4* ldst, const float4* gsrc) {
#if HAVE_ASYNC_LDS
    __builtin_amdgcn_global_load_async_to_lds_b128(
        (__attribute__((address_space(1))) v4i*)gsrc,
        (__attribute__((address_space(3))) v4i*)ldst, 0, 0);
#else
    *ldst = *gsrc;
#endif
}

__device__ __forceinline__ void lds_copy_fence() {
#if HAVE_ASYNC_LDS
    __builtin_amdgcn_s_wait_asynccnt(0);
#endif
}

// ---------------------------------------------------------------------------
// WMMA fragment load for 16-bit A/B operands of v_wmma_f32_16x16x32_f16.
// ISA layout (CDNA5 ISA, 16-bit A 16x32): lane L holds row (L&15);
// lanes 0-15 carry K = {0..7, 16..23}, lanes 16-31 carry K = {8..15, 24..31}.
// Per lane that is two contiguous 16-byte chunks at half-offsets (k0+8g) and
// (k0+8g+16)  ->  two b128 loads.
// ---------------------------------------------------------------------------
union FragU { float4 f4[2]; v16h h; };

__device__ __forceinline__ v16h load_frag16(const _Float16* __restrict__ base,
                                            int ld, int row, int k0, int g) {
    const _Float16* p = base + (size_t)row * ld + k0 + g * 8;
    FragU u;
    u.f4[0] = *(const float4*)(p);
    u.f4[1] = *(const float4*)(p + 16);
    return u.h;
}

// ---------------------------------------------------------------------------
// Generic  C[M,N] = A[M,K] * Bt[N,K]^T  (f16 in, f32 out) via WMMA.
// One wave computes a 16x64 tile: A frag reused across 4 B frags per k-step.
// ---------------------------------------------------------------------------
__global__ void gemm16_wmma(const _Float16* __restrict__ A, int lda,
                            const _Float16* __restrict__ Bt, int ldb,
                            float* __restrict__ C, int ldc,
                            int M, int N, int K) {
    const int lane   = threadIdx.x & 31;
    const int wave   = (int)((blockIdx.x * blockDim.x + threadIdx.x) >> 5);
    const int tilesN = N >> 6;                 // 64-wide N tiles
    const int mi     = wave / tilesN;
    const int ni     = (wave % tilesN) << 6;
    if (mi * 16 >= M) return;

    const int r = lane & 15;
    const int g = lane >> 4;

    v8f acc0 = {}, acc1 = {}, acc2 = {}, acc3 = {};
    for (int k0 = 0; k0 < K; k0 += 32) {
        v16h a  = load_frag16(A,  lda, mi * 16 + r, k0, g);
        v16h b0 = load_frag16(Bt, ldb, ni +  0 + r, k0, g);
        v16h b1 = load_frag16(Bt, ldb, ni + 16 + r, k0, g);
        v16h b2 = load_frag16(Bt, ldb, ni + 32 + r, k0, g);
        v16h b3 = load_frag16(Bt, ldb, ni + 48 + r, k0, g);
        acc0 = __builtin_amdgcn_wmma_f32_16x16x32_f16(false, a, false, b0, (short)0, acc0, false, false);
        acc1 = __builtin_amdgcn_wmma_f32_16x16x32_f16(false, a, false, b1, (short)0, acc1, false, false);
        acc2 = __builtin_amdgcn_wmma_f32_16x16x32_f16(false, a, false, b2, (short)0, acc2, false, false);
        acc3 = __builtin_amdgcn_wmma_f32_16x16x32_f16(false, a, false, b3, (short)0, acc3, false, false);
    }

    // C/D layout: VGPR i -> row (g*8 + i), column (lane&15) within tile.
    #pragma unroll
    for (int i = 0; i < 8; ++i) {
        const size_t row = (size_t)(mi * 16 + g * 8 + i) * ldc + ni + r;
        C[row +  0] = acc0[i];
        C[row + 16] = acc1[i];
        C[row + 32] = acc2[i];
        C[row + 48] = acc3[i];
    }
}

// ---------------------------------------------------------------------------
// Zipf priors: rank by descending freq (stable ties -> lower index first),
// zw = rank^-s, normalize.  One 1024-thread block, freqs staged in LDS.
// ---------------------------------------------------------------------------
__global__ void zipf_priors_kernel(const float* __restrict__ counts,
                                   const float* __restrict__ total,
                                   const float* __restrict__ zs,
                                   float* __restrict__ priors) {
    __shared__ float sf[V_];
    __shared__ float red[32];
    const int t = threadIdx.x;
    const float tot = fmaxf(total[0], 1.0f);
    const float fi  = counts[t] / tot;
    sf[t] = fi;
    __syncthreads();

    int cnt = 0;
    for (int j = 0; j < V_; ++j) {
        const float fj = sf[j];
        cnt += (fj > fi) || (fj == fi && j < t);
    }
    const float rank = (float)(cnt + 1);
    const float zw   = powf(rank, -zs[0]);

    float s = zw;
    for (int off = 16; off; off >>= 1) s += __shfl_xor(s, off, 32);
    if ((t & 31) == 0) red[t >> 5] = s;
    __syncthreads();
    float tsum = 0.0f;
    for (int i = 0; i < 32; ++i) tsum += red[i];
    priors[t] = zw / fmaxf(tsum, 1e-8f);
}

// ---------------------------------------------------------------------------
// Wave-per-row: emit raw-f16 and l2-normalized-f16 copies of F and P rows.
// ---------------------------------------------------------------------------
__global__ void normalize_rows_kernel(const float* __restrict__ feat,
                                      const float* __restrict__ proto,
                                      _Float16* __restrict__ F16,
                                      _Float16* __restrict__ Fn16,
                                      _Float16* __restrict__ P16,
                                      _Float16* __restrict__ Pn16) {
    const int lane = threadIdx.x & 31;
    const int wave = (int)((blockIdx.x * blockDim.x + threadIdx.x) >> 5);
    const float* src; _Float16* raw; _Float16* nrm;
    if (wave < B_) {
        src = feat  + (size_t)wave * D_;
        raw = F16   + (size_t)wave * D_;
        nrm = Fn16  + (size_t)wave * D_;
    } else if (wave < B_ + V_) {
        const int v = wave - B_;
        src = proto + (size_t)v * D_;
        raw = P16   + (size_t)v * D_;
        nrm = Pn16  + (size_t)v * D_;
    } else return;

    float x[16];
    float s = 0.0f;
    #pragma unroll
    for (int k = 0; k < 16; ++k) { x[k] = src[k * 32 + lane]; s += x[k] * x[k]; }
    for (int off = 16; off; off >>= 1) s += __shfl_xor(s, off, 32);
    const float inv = 1.0f / fmaxf(sqrtf(s), 1e-12f);
    #pragma unroll
    for (int k = 0; k < 16; ++k) {
        const int d = k * 32 + lane;
        raw[d] = (_Float16)x[k];
        nrm[d] = (_Float16)(x[k] * inv);
    }
}

__global__ void cvt_f16_kernel(const float* __restrict__ src,
                               _Float16* __restrict__ dst, int n) {
    const int i = blockIdx.x * blockDim.x + threadIdx.x;
    if (i < n) dst[i] = (_Float16)src[i];
}

// ---------------------------------------------------------------------------
// Fused nonlinear part, LDS-tiled.
// Block = 256 threads (8 waves) owns a 16(b) x 64(v) tile of pairs.
//   stage fp-tile (32KB) + pp-tile (128KB) into LDS via async b128 copies,
//   hoist b1/W2 into 16 regs/lane, then one wave per (b,v) pair:
//   logit = sum_d relu(fp+pp+b1)*W2 ; w = priors*(1+sim/temp)*sigmoid(logit+b2)
// Global traffic: 160KB per 1024 pairs (~50x less than the untiled version).
// ---------------------------------------------------------------------------
__global__ void attn_weights_kernel(const float* __restrict__ fp,
                                    const float* __restrict__ pp,
                                    const float* __restrict__ b1v,
                                    const float* __restrict__ W2,
                                    const float* __restrict__ b2v,
                                    const float* __restrict__ temp,
                                    const float* __restrict__ priors,
                                    const float* __restrict__ sim,
                                    float* __restrict__ w) {
    extern __shared__ float smem[];
    float* s_fp = smem;             // 16 * 512 f32
    float* s_pp = smem + 16 * D_;   // 64 * 512 f32

    const int tid  = threadIdx.x;           // 256
    const int lane = tid & 31;
    const int wv   = tid >> 5;              // 0..7
    const int v0   = blockIdx.x * 64;
    const int b0   = blockIdx.y * 16;

    // ---- stage tiles into LDS (async, no VGPR round-trip) ----
    for (int i = tid; i < 16 * (D_ / 4); i += 256) {           // 2048 float4
        const int row = i >> 7, col = i & 127;
        lds_copy16((float4*)s_fp + i,
                   (const float4*)(fp + (size_t)(b0 + row) * D_) + col);
    }
    for (int i = tid; i < 64 * (D_ / 4); i += 256) {           // 8192 float4
        const int row = i >> 7, col = i & 127;
        lds_copy16((float4*)s_pp + i,
                   (const float4*)(pp + (size_t)(v0 + row) * D_) + col);
    }

    // ---- hoist b1 / W2 / scalars into registers while copies fly ----
    float b1r[16], w2r[16];
    #pragma unroll
    for (int k = 0; k < 16; ++k) {
        b1r[k] = b1v[k * 32 + lane];
        w2r[k] = W2[k * 32 + lane];
    }
    const float b2s  = b2v[0];
    const float invT = 1.0f / fmaxf(temp[0], 1e-4f);

    lds_copy_fence();
    __syncthreads();

    // ---- one wave per (b,v) pair; 128 pairs per wave ----
    for (int p = wv; p < 16 * 64; p += 8) {
        const int bl = p >> 6;
        const int vl = p & 63;
        const float* fr = s_fp + bl * D_;
        const float* pr = s_pp + vl * D_;
        float acc = 0.0f;
        #pragma unroll
        for (int k = 0; k < 16; ++k) {
            const int d = k * 32 + lane;
            float x = fr[d] + pr[d] + b1r[k];
            acc = fmaf(fmaxf(x, 0.0f), w2r[k], acc);
        }
        for (int off = 16; off; off >>= 1) acc += __shfl_xor(acc, off, 32);
        if (lane == 0) {
            const int b = b0 + bl, v = v0 + vl;
            const float attn = 1.0f / (1.0f + expf(-(acc + b2s)));
            const float sv   = sim[(size_t)b * V_ + v] * invT;
            w[(size_t)b * V_ + v] = priors[v] * (1.0f + sv) * attn;
        }
    }
}

// ---------------------------------------------------------------------------
// Row softmax over V=1024, one 256-thread block per row (4 elems/thread).
// ---------------------------------------------------------------------------
__global__ void softmax_kernel(const float* __restrict__ w,
                               float* __restrict__ out) {
    __shared__ float red[8];
    const int b = blockIdx.x;
    const int t = threadIdx.x;
    const float* row = w + (size_t)b * V_;

    float x[4];
    float mx = -3.4e38f;
    #pragma unroll
    for (int k = 0; k < 4; ++k) { x[k] = row[t + 256 * k]; mx = fmaxf(mx, x[k]); }
    for (int off = 16; off; off >>= 1) mx = fmaxf(mx, __shfl_xor(mx, off, 32));
    if ((t & 31) == 0) red[t >> 5] = mx;
    __syncthreads();
    float rmx = red[0];
    for (int i = 1; i < 8; ++i) rmx = fmaxf(rmx, red[i]);

    float e[4];
    float s = 0.0f;
    #pragma unroll
    for (int k = 0; k < 4; ++k) { e[k] = expf(x[k] - rmx); s += e[k]; }
    for (int off = 16; off; off >>= 1) s += __shfl_xor(s, off, 32);
    __syncthreads();
    if ((t & 31) == 0) red[t >> 5] = s;
    __syncthreads();
    float tot = 0.0f;
    for (int i = 0; i < 8; ++i) tot += red[i];
    const float inv = 1.0f / tot;
    #pragma unroll
    for (int k = 0; k < 4; ++k) out[(size_t)b * V_ + t + 256 * k] = e[k] * inv;
}

// ---------------------------------------------------------------------------
extern "C" void kernel_launch(void* const* d_in, const int* in_sizes, int n_in,
                              void* d_out, int out_size, void* d_ws, size_t ws_size,
                              hipStream_t stream) {
    (void)in_sizes; (void)n_in; (void)out_size; (void)ws_size;
    const float* features   = (const float*)d_in[0];   // (B, D)
    const float* counts     = (const float*)d_in[1];   // (V,)
    const float* total      = (const float*)d_in[2];   // scalar
    const float* protos     = (const float*)d_in[3];   // (V, D)
    const float* zipf_s     = (const float*)d_in[4];   // scalar
    const float* temp       = (const float*)d_in[5];   // scalar
    const float* W1         = (const float*)d_in[6];   // (D, 2D)
    const float* b1v        = (const float*)d_in[7];   // (D,)
    const float* W2         = (const float*)d_in[8];   // (1, D)
    const float* b2v        = (const float*)d_in[9];   // scalar
    float* out = (float*)d_out;                        // (B, V)

    // Workspace carve-up (all 256B aligned)
    char* ws = (char*)d_ws;
    size_t off = 0;
    auto carve = [&](size_t bytes) { void* p = ws + off; off = (off + bytes + 255) & ~(size_t)255; return p; };
    float*    priors = (float*)   carve(V_ * sizeof(float));
    _Float16* W1h    = (_Float16*)carve((size_t)D_ * 2 * D_ * sizeof(_Float16));
    _Float16* F16    = (_Float16*)carve((size_t)B_ * D_ * sizeof(_Float16));
    _Float16* Fn16   = (_Float16*)carve((size_t)B_ * D_ * sizeof(_Float16));
    _Float16* P16    = (_Float16*)carve((size_t)V_ * D_ * sizeof(_Float16));
    _Float16* Pn16   = (_Float16*)carve((size_t)V_ * D_ * sizeof(_Float16));
    float*    fp     = (float*)   carve((size_t)B_ * D_ * sizeof(float));
    float*    pp     = (float*)   carve((size_t)V_ * D_ * sizeof(float));
    float*    sim    = (float*)   carve((size_t)B_ * V_ * sizeof(float));
    float*    wbuf   = (float*)   carve((size_t)B_ * V_ * sizeof(float));

    // 1) Zipf priors (1 block, V threads)
    zipf_priors_kernel<<<1, V_, 0, stream>>>(counts, total, zipf_s, priors);

    // 2) f16 conversions
    {
        const int n = D_ * 2 * D_;
        cvt_f16_kernel<<<(n + 255) / 256, 256, 0, stream>>>(W1, W1h, n);
    }
    {
        const int waves = B_ + V_;                       // one wave per row
        normalize_rows_kernel<<<(waves * 32 + 255) / 256, 256, 0, stream>>>(
            features, protos, F16, Fn16, P16, Pn16);
    }

    // 3) GEMMs via v_wmma_f32_16x16x32_f16 (16x64 tile per wave)
    // sim = Fn @ Pn^T          : M=B, N=V, K=D
    gemm16_wmma<<<(B_ / 16) * (V_ / 64) * 32 / 256, 256, 0, stream>>>(
        Fn16, D_, Pn16, D_, sim, V_, B_, V_, D_);
    // fp = F @ W1[:, :D]^T     : Bt row i = W1h row i (ld = 2D)
    gemm16_wmma<<<(B_ / 16) * (D_ / 64) * 32 / 256, 256, 0, stream>>>(
        F16, D_, W1h, 2 * D_, fp, D_, B_, D_, D_);
    // pp = P @ W1[:, D:]^T     : Bt row i = W1h row i + D offset
    gemm16_wmma<<<(V_ / 16) * (D_ / 64) * 32 / 256, 256, 0, stream>>>(
        P16, D_, W1h + D_, 2 * D_, pp, D_, V_, D_, D_);

    // 4) Fused relu/attn/weights: LDS-tiled 16x64 pair blocks (160KB LDS/WG)
    {
        const size_t shbytes = (size_t)(16 + 64) * D_ * sizeof(float);
        attn_weights_kernel<<<dim3(V_ / 64, B_ / 16), 256, shbytes, stream>>>(
            fp, pp, b1v, W2, b2v, temp, priors, sim, wbuf);
    }

    // 5) Row softmax -> output
    softmax_kernel<<<B_, 256, 0, stream>>>(wbuf, out);
}
